// LSTM_65352222376428
// MI455X (gfx1250) — matrix operand order repaired and checked by reading, not compiled, not run on hardware
//
#include <hip/hip_runtime.h>
#include <hip/hip_bf16.h>
#include <math.h>

// ---------------------------------------------------------------------------
// LSTM on gfx1250 (MI455X), bf16 WMMA 16x16x32 with fp32 accumulate.
//   S=512 steps, B=64 batch, I=512 in, H=1024 hidden.
//   Per step: G[64,4096] = [h_{t-1} | x_t] (64x1536) @ Wpack^T (1536x4096) + bias
//   then pointwise LSTM cell update, fused in one kernel.
// ---------------------------------------------------------------------------

typedef __attribute__((ext_vector_type(16))) __bf16 v16bf;
typedef __attribute__((ext_vector_type(8)))  __bf16 v8bf;
typedef __attribute__((ext_vector_type(8)))  float  v8f;

#define S_  512
#define B_  64
#define I_  512
#define H_  1024
#define KT  1536                 // H + I  (fused K dimension)
#define NT  4096                 // 4 gates * H
#define APAD 8
#define ASTRIDE (KT + APAD)      // 1544 bf16 -> 3088 B row stride (16B-aligned, bank-staggered)

// ---------------------------------------------------------------------------
// Pack recurrent+input weights of all 4 gates into one bf16 matrix:
//   Wpack[g*1024 + h][0:1024]    = Wgh[h][:]
//   Wpack[g*1024 + h][1024:1536] = Wgx[h][:]
// Gate order: 0=f, 1=i, 2=o, 3=c(candidate)
// ---------------------------------------------------------------------------
__global__ __launch_bounds__(256)
void lstm_pack_weights(const float* __restrict__ Wfh, const float* __restrict__ Wih,
                       const float* __restrict__ Woh, const float* __restrict__ Wch,
                       const float* __restrict__ Wfx, const float* __restrict__ Wix,
                       const float* __restrict__ Wox, const float* __restrict__ Wcx,
                       __bf16* __restrict__ Wpack)
{
    const int r = blockIdx.x;          // 0..4095 (gate-major row)
    const int g = r >> 10;
    const int h = r & 1023;
    const float* Wh = (g == 0) ? Wfh : (g == 1) ? Wih : (g == 2) ? Woh : Wch;
    const float* Wx = (g == 0) ? Wfx : (g == 1) ? Wix : (g == 2) ? Wox : Wcx;
    __bf16* dst = Wpack + (size_t)r * KT;
    for (int j = threadIdx.x; j < KT; j += 256) {
        float v = (j < H_) ? Wh[(size_t)h * H_ + j] : Wx[(size_t)h * I_ + (j - H_)];
        dst[j] = (__bf16)v;
    }
}

// ---------------------------------------------------------------------------
// Initialize h/c state from inputs (h0 = h.sum(0) == h[0] since dim0==1),
// and fold the two bias vectors per gate into one: bias[g*1024+h] = bgx[h]+bgh[h].
// ---------------------------------------------------------------------------
__global__ __launch_bounds__(256)
void lstm_init(const float* __restrict__ h0, const float* __restrict__ c0,
               const float* __restrict__ bfx, const float* __restrict__ bfh,
               const float* __restrict__ bix, const float* __restrict__ bih,
               const float* __restrict__ box, const float* __restrict__ boh,
               const float* __restrict__ bcx, const float* __restrict__ bch,
               float* __restrict__ hstA, float* __restrict__ hstB,
               float* __restrict__ cst,  float* __restrict__ bias)
{
    const int idx = blockIdx.x * 256 + threadIdx.x;     // 0..65535
    if (idx < B_ * H_) {
        hstA[idx] = h0[idx];
        hstB[idx] = 0.0f;
        cst[idx]  = c0[idx];
    }
    if (idx < NT) {
        const int g = idx >> 10;
        const int h = idx & 1023;
        const float* bx = (g == 0) ? bfx : (g == 1) ? bix : (g == 2) ? box : bcx;
        const float* bh = (g == 0) ? bfh : (g == 1) ? bih : (g == 2) ? boh : bch;
        bias[idx] = bx[h] + bh[h];
    }
}

// ---------------------------------------------------------------------------
// One LSTM step. Grid: (32 h-groups of 32 cols, 4 batch tiles of 16 rows).
// Block: 256 threads = 8 waves. Wave w -> gate g = w>>1, subtile = w&1.
// Each wave computes one 16x16 tile of the pre-activation GEMM with
// v_wmma_f32_16x16x32_bf16, K = 1536 (h then x), then the block fuses the
// pointwise cell update through an LDS gate buffer.
// ---------------------------------------------------------------------------
__global__ __launch_bounds__(256)
void lstm_step(const float* __restrict__ x,
               const __bf16* __restrict__ Wpack,
               const float* __restrict__ bias,
               const float* __restrict__ hprev,
               float* __restrict__ hnext,
               float* __restrict__ cstate,
               float* __restrict__ hseq,
               float* __restrict__ hlast,
               float* __restrict__ clast,
               int t, int is_last)
{
    __shared__ __align__(32) __bf16 Alds[16][ASTRIDE];   // 16 x 1536 bf16 A tile (padded)
    __shared__ float gbuf[4][2][16][16];                 // pre-activations: [gate][sub][m][n]

    const int tid    = threadIdx.x;
    const int hgroup = blockIdx.x;       // 0..31  -> 32 hidden columns each
    const int mtile  = blockIdx.y;       // 0..3   -> 16 batch rows each
    const int brow0  = mtile * 16;

    const float* xt = x + (size_t)t * B_ * I_;

    // ---- Stage A = [h_{t-1} | x_t] rows brow0..brow0+15, fp32 -> bf16 in LDS
    for (int e = tid; e < 16 * KT; e += 256) {
        const int row = e / KT;
        const int col = e - row * KT;
        const int b   = brow0 + row;
        const float v = (col < H_) ? hprev[(size_t)b * H_ + col]
                                   : xt[(size_t)b * I_ + (col - H_)];
        Alds[row][col] = (__bf16)v;
    }
    __syncthreads();

    // ---- Wave tile assignment
    const int wave = tid >> 5;               // 0..7
    const int lane = tid & 31;
    const int g    = wave >> 1;              // gate 0..3
    const int sub  = wave & 1;               // 16-col subtile
    const int n0   = g * H_ + hgroup * 32 + sub * 16;   // global Wpack row / gate col

    // A-fragment addressing (ISA 7.12.2, 16-bit A 16x32):
    //   lanes 0-15: row=lane,   K halves 0..7 -> K 0..7,  8..15 -> K 16..23
    //   lanes16-31: row=lane-16,K halves 0..7 -> K 8..15, 8..15 -> K 24..31
    const int arow  = lane & 15;
    const int kbase = (lane >> 4) << 3;      // 0 or 8
    // B-fragment addressing (K x 16, col per lane), same K order as A:
    //   lanes 0-15: col=lane,    K 0..7, 16..23 ; lanes 16-31: col=lane-16, K 8..15, 24..31
    const int nl    = lane & 15;
    const int bkoff = (lane >> 4) << 3;      // 0 or 8

    const __bf16* brow_ptr = Wpack + (size_t)(n0 + nl) * KT;

    v8f acc = {};
    for (int kk = 0; kk < KT; kk += 32) {
        union { v16bf v; v8bf h[2]; } a;
        a.h[0] = *(const v8bf*)(&Alds[arow][kk + kbase]);        // K kk+kbase .. +7
        a.h[1] = *(const v8bf*)(&Alds[arow][kk + 16 + kbase]);   // K kk+16+kbase .. +7
        union { v16bf v; v8bf h[2]; } bu;
        bu.h[0] = *(const v8bf*)(brow_ptr + kk + bkoff);
        bu.h[1] = *(const v8bf*)(brow_ptr + kk + 16 + bkoff);
        const v16bf bfrag = bu.v;
        acc = __builtin_amdgcn_wmma_f32_16x16x32_bf16(
                  /*neg_a=*/false, a.v, /*neg_b=*/false, bfrag,
                  /*c_mod=*/(short)0, acc, /*reuse_a=*/false, /*reuse_b=*/false);
    }

    // ---- Epilogue: add fused bias, spill tile to LDS gate buffer.
    // C layout: VGPR r -> (M = r + 8*(lane>=16), N = lane&15)
    const float bv = bias[n0 + nl];
    const int m0 = (lane >> 4) << 3;
    #pragma unroll
    for (int r = 0; r < 8; ++r)
        gbuf[g][sub][m0 + r][nl] = acc[r] + bv;

    __syncthreads();

    // ---- Pointwise LSTM cell update: 16 rows x 32 cols = 512 elems, 2/thread
    for (int e = tid; e < 16 * 32; e += 256) {
        const int m  = e >> 5;
        const int nc = e & 31;
        const int su = nc >> 4;
        const int n  = nc & 15;
        const float pf = gbuf[0][su][m][n];
        const float pi = gbuf[1][su][m][n];
        const float po = gbuf[2][su][m][n];
        const float pc = gbuf[3][su][m][n];
        const float fg = 1.0f / (1.0f + __expf(-pf));
        const float ig = 1.0f / (1.0f + __expf(-pi));
        const float og = 1.0f / (1.0f + __expf(-po));
        const int   b  = brow0 + m;
        const int   hc = hgroup * 32 + nc;
        const size_t idx = (size_t)b * H_ + hc;
        const float cp = cstate[idx];
        const float cn = fg * cp + ig * tanhf(pc);
        const float hn = og * tanhf(cn);
        cstate[idx] = cn;
        hnext[idx]  = hn;
        hseq[((size_t)t * B_ + b) * H_ + hc] = hn;
        if (is_last) { hlast[idx] = hn; clast[idx] = cn; }
    }
}

// ---------------------------------------------------------------------------
extern "C" void kernel_launch(void* const* d_in, const int* in_sizes, int n_in,
                              void* d_out, int out_size, void* d_ws, size_t ws_size,
                              hipStream_t stream)
{
    const float* x   = (const float*)d_in[0];
    const float* h0  = (const float*)d_in[1];
    const float* c0  = (const float*)d_in[2];
    const float* Wfx = (const float*)d_in[3];
    const float* bfx = (const float*)d_in[4];
    const float* Wfh = (const float*)d_in[5];
    const float* bfh = (const float*)d_in[6];
    const float* Wix = (const float*)d_in[7];
    const float* bix = (const float*)d_in[8];
    const float* Wih = (const float*)d_in[9];
    const float* bih = (const float*)d_in[10];
    const float* Wox = (const float*)d_in[11];
    const float* box = (const float*)d_in[12];
    const float* Woh = (const float*)d_in[13];
    const float* boh = (const float*)d_in[14];
    const float* Wcx = (const float*)d_in[15];
    const float* bcx = (const float*)d_in[16];
    const float* Wch = (const float*)d_in[17];
    const float* bch = (const float*)d_in[18];

    // ---- Workspace layout (all offsets 256B-aligned), total ~13.4 MB
    char* ws = (char*)d_ws;
    __bf16* Wpack = (__bf16*)(ws + 0);               // 4096*1536*2 = 12,582,912 B
    float*  bias  = (float*)(ws + 12582912);         //      16,384 B
    float*  hstA  = (float*)(ws + 12599296);         //     262,144 B
    float*  hstB  = (float*)(ws + 12861440);         //     262,144 B
    float*  cst   = (float*)(ws + 13123584);         //     262,144 B

    float* hseq  = (float*)d_out;                    // [S,B,H]
    float* hlast = hseq + (size_t)S_ * B_ * H_;      // [B,H]
    float* clast = hlast + (size_t)B_ * H_;          // [B,H]

    // ---- Prologue: pack weights to bf16, init states + fused bias
    lstm_pack_weights<<<NT, 256, 0, stream>>>(Wfh, Wih, Woh, Wch,
                                              Wfx, Wix, Wox, Wcx, Wpack);
    lstm_init<<<(B_ * H_) / 256, 256, 0, stream>>>(h0, c0,
                                                   bfx, bfh, bix, bih,
                                                   box, boh, bcx, bch,
                                                   hstA, hstB, cst, bias);

    // ---- Recurrence: 512 sequential steps, ping-pong h state
    dim3 grid(H_ / 32, B_ / 16);   // (32, 4)
    for (int t = 0; t < S_; ++t) {
        float* hp = (t & 1) ? hstB : hstA;
        float* hn = (t & 1) ? hstA : hstB;
        lstm_step<<<grid, 256, 0, stream>>>(x, Wpack, bias, hp, hn, cst,
                                            hseq, hlast, clast,
                                            t, (t == S_ - 1) ? 1 : 0);
    }
}
